// MultiHeadAttention_40286793237173
// MI455X (gfx1250) — matrix operand order repaired and checked by reading, not compiled, not run on hardware
//
#include <hip/hip_runtime.h>

#define DI __device__ __forceinline__

typedef __attribute__((ext_vector_type(16))) __bf16 v16bf;
typedef __attribute__((ext_vector_type(8)))  float  v8f;
typedef unsigned int   u32;
typedef unsigned short u16;

struct U2 { u32 x, y; };
union Frag { v16bf bf; U2 q[4]; };

constexpr int BATCH = 8, T = 1024, E = 1024, H = 16, HS = 64;
constexpr size_t SZ_X = (size_t)BATCH * T * E;     // 8388608
constexpr size_t SZ_W = (size_t)H * E * HS;        // 1048576
constexpr size_t SZ_P = (size_t)E * E;             // 1048576
constexpr size_t SZ_Q = (size_t)BATCH * H * T * HS;// 8388608

// workspace offsets (units: ushort/bf16 elements)
constexpr size_t OFF_XB  = 0;
constexpr size_t OFF_WQT = OFF_XB  + SZ_X;
constexpr size_t OFF_WKT = OFF_WQT + SZ_W;
constexpr size_t OFF_WVT = OFF_WKT + SZ_W;
constexpr size_t OFF_WPB = OFF_WVT + SZ_W;
constexpr size_t OFF_QB  = OFF_WPB + SZ_P;
constexpr size_t OFF_KB  = OFF_QB  + SZ_Q;
constexpr size_t OFF_VTB = OFF_KB  + SZ_Q;
constexpr size_t OFF_YB  = OFF_VTB + SZ_Q;

DI u16 f2bf(float f) {
  union { float f; u32 u; } x; x.f = f;
  u32 r = x.u + 0x7FFFu + ((x.u >> 16) & 1u);
  return (u16)(r >> 16);
}

DI v8f wmma_bf16(const Frag& a, const Frag& b, v8f c) {
  return __builtin_amdgcn_wmma_f32_16x16x32_bf16(false, a.bf, false, b.bf,
                                                 (short)0, c, false, false);
}

// ---------------- conversion kernels ----------------
__global__ void k_cvt(const float* __restrict__ in, u16* __restrict__ out, int n) {
  int i = blockIdx.x * blockDim.x + threadIdx.x;
  if (i < n) out[i] = f2bf(in[i]);
}

// Wq/Wk/Wv [H][E][HS] -> bf16 transposed [H][HS][E] (WMMA-B gather layout)
__global__ void k_cvtT(const float* __restrict__ in, u16* __restrict__ out) {
  int idx = blockIdx.x * blockDim.x + threadIdx.x;
  if (idx < (int)SZ_W) {
    int d = idx % HS;
    int e = (idx / HS) % E;
    int h = idx / (HS * E);
    out[((size_t)h * HS + d) * E + e] = f2bf(in[idx]);
  }
}

// ---------------- QKV projection GEMM ----------------
// block: one (b,h), 128 T-rows x 64 hs-cols; double-buffered K-loop over E.
// wave w owns row strip w*16..w*16+15 and all 4 col tiles (1 A frag -> 4 WMMA).
__global__ __launch_bounds__(256) void k_qkv(const u16* __restrict__ xb,
                                             const u16* __restrict__ Wt,
                                             u16* __restrict__ out,
                                             int storeT) {
  __shared__ u16 sA[2][128 * 40];
  __shared__ u16 sB[2][64 * 40];

  const int t0   = blockIdx.x * 128;
  const int bh   = blockIdx.y;           // b*H + h
  const int b    = bh >> 4;
  const int h    = bh & 15;
  const int lane = threadIdx.x & 31;
  const int wave = threadIdx.x >> 5;     // row strip
  const int rlo  = lane & 15;
  const int hs_  = lane >> 4;

  v8f acc[4];
#pragma unroll
  for (int n = 0; n < 4; ++n) acc[n] = v8f{0,0,0,0,0,0,0,0};

  // loader geometry
  const int ra0 = threadIdx.x >> 2;             // A rows 0..63 (chunk 0)
  const int ra1 = ra0 + 64;                     // A rows 64..127 (chunk 1)
  const int ca  = (threadIdx.x & 3) * 8;
  const size_t gA0 = ((size_t)(b * T + t0 + ra0)) * E + ca;
  const size_t gA1 = ((size_t)(b * T + t0 + ra1)) * E + ca;
  const size_t gB0 = ((size_t)(h * HS + ra0)) * E + ca;

  // preload tile 0 into buffer 0
  {
    uint4 a0 = *(const uint4*)(xb + gA0);
    uint4 a1 = *(const uint4*)(xb + gA1);
    uint4 b0 = *(const uint4*)(Wt + gB0);
    *(U2*)&sA[0][ra0 * 40 + ca]     = U2{a0.x, a0.y};
    *(U2*)&sA[0][ra0 * 40 + ca + 4] = U2{a0.z, a0.w};
    *(U2*)&sA[0][ra1 * 40 + ca]     = U2{a1.x, a1.y};
    *(U2*)&sA[0][ra1 * 40 + ca + 4] = U2{a1.z, a1.w};
    *(U2*)&sB[0][ra0 * 40 + ca]     = U2{b0.x, b0.y};
    *(U2*)&sB[0][ra0 * 40 + ca + 4] = U2{b0.z, b0.w};
  }

  for (int it = 0; it < E / 32; ++it) {
    const int cur = it & 1, nxt = cur ^ 1;
    __syncthreads();   // buf[cur] fully staged; buf[nxt] free to overwrite

    // issue next tile's global loads before the matrix math (latency hiding)
    uint4 pa0, pa1, pb0;
    const bool have = (it + 1) < (E / 32);
    if (have) {
      const size_t o = (size_t)(it + 1) * 32;
      pa0 = *(const uint4*)(xb + gA0 + o);
      pa1 = *(const uint4*)(xb + gA1 + o);
      pb0 = *(const uint4*)(Wt + gB0 + o);
    }

    // compute from buf[cur]
    Frag a;
    {
      const int base = (wave * 16 + rlo) * 40 + hs_ * 8;
      a.q[0] = *(const U2*)&sA[cur][base];
      a.q[1] = *(const U2*)&sA[cur][base + 4];
      a.q[2] = *(const U2*)&sA[cur][base + 16];
      a.q[3] = *(const U2*)&sA[cur][base + 20];
    }
#pragma unroll
    for (int n = 0; n < 4; ++n) {
      Frag bfr;
      const int base = (n * 16 + rlo) * 40 + hs_ * 8;
      bfr.q[0] = *(const U2*)&sB[cur][base];
      bfr.q[1] = *(const U2*)&sB[cur][base + 4];
      bfr.q[2] = *(const U2*)&sB[cur][base + 16];
      bfr.q[3] = *(const U2*)&sB[cur][base + 20];
      acc[n] = wmma_bf16(a, bfr, acc[n]);
    }

    // stage next tile into buf[nxt]
    if (have) {
      *(U2*)&sA[nxt][ra0 * 40 + ca]     = U2{pa0.x, pa0.y};
      *(U2*)&sA[nxt][ra0 * 40 + ca + 4] = U2{pa0.z, pa0.w};
      *(U2*)&sA[nxt][ra1 * 40 + ca]     = U2{pa1.x, pa1.y};
      *(U2*)&sA[nxt][ra1 * 40 + ca + 4] = U2{pa1.z, pa1.w};
      *(U2*)&sB[nxt][ra0 * 40 + ca]     = U2{pb0.x, pb0.y};
      *(U2*)&sB[nxt][ra0 * 40 + ca + 4] = U2{pb0.z, pb0.w};
    }
  }

  // store bf16: normal [bh][t][d] or transposed [bh][d][t] (for V)
#pragma unroll
  for (int n = 0; n < 4; ++n) {
    const int col = n * 16 + rlo;
#pragma unroll
    for (int vi = 0; vi < 8; ++vi) {
      const int t = t0 + wave * 16 + vi + 8 * hs_;
      const u16 v = f2bf(acc[n][vi]);
      if (storeT)
        out[((size_t)bh * HS + col) * T + t] = v;
      else
        out[((size_t)bh * T + t) * HS + col] = v;
    }
  }
}

// ---------------- flash-style causal attention ----------------
// block: one (b,h), 128 queries (8 waves x 16-row strips); key blocks of 64
__global__ __launch_bounds__(256) void k_attn(const u16* __restrict__ qb,
                                              const u16* __restrict__ kb,
                                              const u16* __restrict__ vtb,
                                              u16* __restrict__ yb) {
  __shared__ u16 sK[64 * 72];            // [s][d] gather layout for QK^T
  __shared__ u16 sV[64 * 72];            // [d][s] gather layout for PV
  __shared__ u16 sP[8 * 16 * 72];        // per-wave P round-trip buffer

  const int bh    = blockIdx.y;
  const int b     = bh >> 4;
  const int h     = bh & 15;
  const int tBase = blockIdx.x * 128;
  const int lane  = threadIdx.x & 31;
  const int wave  = threadIdx.x >> 5;
  const int rlo   = lane & 15;
  const int hs_   = lane >> 4;
  const int t0w   = tBase + wave * 16;

  // Q fragments straight from global (A layout, hs split into 2 K=32 frags)
  Frag qf[2];
  {
    const u16* qrow = qb + ((size_t)bh * T + t0w + rlo) * HS;
#pragma unroll
    for (int kf = 0; kf < 2; ++kf) {
      const int base = kf * 32 + hs_ * 8;
      qf[kf].q[0] = *(const U2*)(qrow + base);
      qf[kf].q[1] = *(const U2*)(qrow + base + 4);
      qf[kf].q[2] = *(const U2*)(qrow + base + 16);
      qf[kf].q[3] = *(const U2*)(qrow + base + 20);
    }
  }

  v8f accO[4];
#pragma unroll
  for (int n = 0; n < 4; ++n) accO[n] = v8f{0,0,0,0,0,0,0,0};
  float m[8], l[8];
#pragma unroll
  for (int vi = 0; vi < 8; ++vi) { m[vi] = -1e30f; l[vi] = 0.f; }

  u16* pbuf = sP + wave * 16 * 72;
  const int nIter = (tBase >> 6) + 2;

  for (int it = 0; it < nIter; ++it) {
    const int s0 = it * 64;
    // stage K [s][d] and V^T [d][s] tiles
#pragma unroll
    for (int i = 0; i < 2; ++i) {
      const int idx = threadIdx.x + i * 256;
      const int r = idx >> 3, c = (idx & 7) * 8;
      uint4 kv = *(const uint4*)(kb + ((size_t)bh * T + s0 + r) * HS + c);
      *(U2*)&sK[r * 72 + c]     = U2{kv.x, kv.y};
      *(U2*)&sK[r * 72 + c + 4] = U2{kv.z, kv.w};
      uint4 vv = *(const uint4*)(vtb + ((size_t)bh * HS + r) * T + s0 + c);
      *(U2*)&sV[r * 72 + c]     = U2{vv.x, vv.y};
      *(U2*)&sV[r * 72 + c + 4] = U2{vv.z, vv.w};
    }
    __syncthreads();

    // S = Q K^T  (4 col tiles x K=64)
    v8f S[4];
#pragma unroll
    for (int n = 0; n < 4; ++n) {
      v8f s = v8f{0,0,0,0,0,0,0,0};
#pragma unroll
      for (int kf = 0; kf < 2; ++kf) {
        Frag kfr;
        const int base = (n * 16 + rlo) * 72 + kf * 32 + hs_ * 8;
        kfr.q[0] = *(const U2*)&sK[base];
        kfr.q[1] = *(const U2*)&sK[base + 4];
        kfr.q[2] = *(const U2*)&sK[base + 16];
        kfr.q[3] = *(const U2*)&sK[base + 20];
        s = wmma_bf16(qf[kf], kfr, s);
      }
      S[n] = s;
    }

    // scale + causal mask + row max
    float rm[8];
#pragma unroll
    for (int vi = 0; vi < 8; ++vi) rm[vi] = -1e30f;
#pragma unroll
    for (int n = 0; n < 4; ++n) {
      const int scol = s0 + n * 16 + rlo;
#pragma unroll
      for (int vi = 0; vi < 8; ++vi) {
        const int t = t0w + vi + 8 * hs_;
        float v = S[n][vi] * 0.125f;
        if (scol > t) v = -1e30f;
        S[n][vi] = v;
        rm[vi] = fmaxf(rm[vi], v);
      }
    }
#pragma unroll
    for (int off = 1; off < 16; off <<= 1)
#pragma unroll
      for (int vi = 0; vi < 8; ++vi)
        rm[vi] = fmaxf(rm[vi], __shfl_xor(rm[vi], off, 32));

    // online softmax update
    float corr[8];
#pragma unroll
    for (int vi = 0; vi < 8; ++vi) {
      const float mn = fmaxf(m[vi], rm[vi]);
      corr[vi] = __expf(m[vi] - mn);
      m[vi] = mn;
      l[vi] *= corr[vi];
    }
#pragma unroll
    for (int n = 0; n < 4; ++n)
#pragma unroll
      for (int vi = 0; vi < 8; ++vi) accO[n][vi] *= corr[vi];

    float rs[8];
#pragma unroll
    for (int vi = 0; vi < 8; ++vi) rs[vi] = 0.f;
#pragma unroll
    for (int n = 0; n < 4; ++n)
#pragma unroll
      for (int vi = 0; vi < 8; ++vi) {
        const float p = __expf(S[n][vi] - m[vi]);
        rs[vi] += p;
        pbuf[(vi + 8 * hs_) * 72 + n * 16 + rlo] = f2bf(p);  // C->A relayout via LDS
      }
#pragma unroll
    for (int off = 1; off < 16; off <<= 1)
#pragma unroll
      for (int vi = 0; vi < 8; ++vi) rs[vi] += __shfl_xor(rs[vi], off, 32);
#pragma unroll
    for (int vi = 0; vi < 8; ++vi) l[vi] += rs[vi];

    // reload P as A fragments (wave-private buffer; LDS is in-order per wave)
    Frag pf[2];
#pragma unroll
    for (int kf = 0; kf < 2; ++kf) {
      const int base = rlo * 72 + kf * 32 + hs_ * 8;
      pf[kf].q[0] = *(const U2*)&pbuf[base];
      pf[kf].q[1] = *(const U2*)&pbuf[base + 4];
      pf[kf].q[2] = *(const U2*)&pbuf[base + 16];
      pf[kf].q[3] = *(const U2*)&pbuf[base + 20];
    }

    // O += P V
#pragma unroll
    for (int n = 0; n < 4; ++n)
#pragma unroll
      for (int kf = 0; kf < 2; ++kf) {
        Frag vfr;
        const int base = (n * 16 + rlo) * 72 + kf * 32 + hs_ * 8;
        vfr.q[0] = *(const U2*)&sV[base];
        vfr.q[1] = *(const U2*)&sV[base + 4];
        vfr.q[2] = *(const U2*)&sV[base + 16];
        vfr.q[3] = *(const U2*)&sV[base + 20];
        accO[n] = wmma_bf16(pf[kf], vfr, accO[n]);
      }
    __syncthreads();
  }

  // normalize and write concat-head layout y[b, t, h*64 + d] (bf16)
#pragma unroll
  for (int n = 0; n < 4; ++n) {
    const int col = h * HS + n * 16 + rlo;
#pragma unroll
    for (int vi = 0; vi < 8; ++vi) {
      const int t = t0w + vi + 8 * hs_;
      yb[((size_t)b * T + t) * E + col] = f2bf(accO[n][vi] / l[vi]);
    }
  }
}

// ---------------- output projection: out = y @ Wp^T + bp (fp32 out) ----------
// block: 128 rows x 64 cols, double-buffered; wave owns 1 row strip x 4 col tiles
__global__ __launch_bounds__(256) void k_proj(const u16* __restrict__ yb,
                                              const u16* __restrict__ Wpb,
                                              const float* __restrict__ bp,
                                              float* __restrict__ out) {
  __shared__ u16 sA[2][128 * 40];
  __shared__ u16 sB[2][64 * 40];

  const int r0   = blockIdx.x * 128;  // rows over B*T
  const int i0   = blockIdx.y * 64;   // output feature cols
  const int lane = threadIdx.x & 31;
  const int wave = threadIdx.x >> 5;
  const int rlo  = lane & 15;
  const int hs_  = lane >> 4;

  v8f acc[4];
#pragma unroll
  for (int n = 0; n < 4; ++n) acc[n] = v8f{0,0,0,0,0,0,0,0};

  const int ra0 = threadIdx.x >> 2;
  const int ra1 = ra0 + 64;
  const int ca  = (threadIdx.x & 3) * 8;
  const size_t gA0 = ((size_t)(r0 + ra0)) * E + ca;
  const size_t gA1 = ((size_t)(r0 + ra1)) * E + ca;
  const size_t gB0 = ((size_t)(i0 + ra0)) * E + ca;  // Wp already in gather layout

  {
    uint4 a0 = *(const uint4*)(yb + gA0);
    uint4 a1 = *(const uint4*)(yb + gA1);
    uint4 b0 = *(const uint4*)(Wpb + gB0);
    *(U2*)&sA[0][ra0 * 40 + ca]     = U2{a0.x, a0.y};
    *(U2*)&sA[0][ra0 * 40 + ca + 4] = U2{a0.z, a0.w};
    *(U2*)&sA[0][ra1 * 40 + ca]     = U2{a1.x, a1.y};
    *(U2*)&sA[0][ra1 * 40 + ca + 4] = U2{a1.z, a1.w};
    *(U2*)&sB[0][ra0 * 40 + ca]     = U2{b0.x, b0.y};
    *(U2*)&sB[0][ra0 * 40 + ca + 4] = U2{b0.z, b0.w};
  }

  for (int it = 0; it < E / 32; ++it) {
    const int cur = it & 1, nxt = cur ^ 1;
    __syncthreads();

    uint4 pa0, pa1, pb0;
    const bool have = (it + 1) < (E / 32);
    if (have) {
      const size_t o = (size_t)(it + 1) * 32;
      pa0 = *(const uint4*)(yb + gA0 + o);
      pa1 = *(const uint4*)(yb + gA1 + o);
      pb0 = *(const uint4*)(Wpb + gB0 + o);
    }

    Frag a;
    {
      const int base = (wave * 16 + rlo) * 40 + hs_ * 8;
      a.q[0] = *(const U2*)&sA[cur][base];
      a.q[1] = *(const U2*)&sA[cur][base + 4];
      a.q[2] = *(const U2*)&sA[cur][base + 16];
      a.q[3] = *(const U2*)&sA[cur][base + 20];
    }
#pragma unroll
    for (int n = 0; n < 4; ++n) {
      Frag bfr;
      const int base = (n * 16 + rlo) * 40 + hs_ * 8;
      bfr.q[0] = *(const U2*)&sB[cur][base];
      bfr.q[1] = *(const U2*)&sB[cur][base + 4];
      bfr.q[2] = *(const U2*)&sB[cur][base + 16];
      bfr.q[3] = *(const U2*)&sB[cur][base + 20];
      acc[n] = wmma_bf16(a, bfr, acc[n]);
    }

    if (have) {
      *(U2*)&sA[nxt][ra0 * 40 + ca]     = U2{pa0.x, pa0.y};
      *(U2*)&sA[nxt][ra0 * 40 + ca + 4] = U2{pa0.z, pa0.w};
      *(U2*)&sA[nxt][ra1 * 40 + ca]     = U2{pa1.x, pa1.y};
      *(U2*)&sA[nxt][ra1 * 40 + ca + 4] = U2{pa1.z, pa1.w};
      *(U2*)&sB[nxt][ra0 * 40 + ca]     = U2{pb0.x, pb0.y};
      *(U2*)&sB[nxt][ra0 * 40 + ca + 4] = U2{pb0.z, pb0.w};
    }
  }

#pragma unroll
  for (int n = 0; n < 4; ++n) {
    const int col = i0 + n * 16 + rlo;
    const float bias = bp[col];
#pragma unroll
    for (int vi = 0; vi < 8; ++vi) {
      const int row = r0 + wave * 16 + vi + 8 * hs_;
      out[(size_t)row * E + col] = acc[n][vi] + bias;
    }
  }
}

// ---------------- launcher ----------------
extern "C" void kernel_launch(void* const* d_in, const int* in_sizes, int n_in,
                              void* d_out, int out_size, void* d_ws, size_t ws_size,
                              hipStream_t stream) {
  (void)in_sizes; (void)n_in; (void)out_size; (void)ws_size;
  const float* x  = (const float*)d_in[0];
  const float* Wq = (const float*)d_in[1];
  const float* Wk = (const float*)d_in[2];
  const float* Wv = (const float*)d_in[3];
  const float* Wp = (const float*)d_in[4];
  const float* bp = (const float*)d_in[5];
  u16* ws = (u16*)d_ws;
  float* out = (float*)d_out;

  k_cvt <<<(int)((SZ_X + 255) / 256), 256, 0, stream>>>(x,  ws + OFF_XB,  (int)SZ_X);
  k_cvtT<<<(int)((SZ_W + 255) / 256), 256, 0, stream>>>(Wq, ws + OFF_WQT);
  k_cvtT<<<(int)((SZ_W + 255) / 256), 256, 0, stream>>>(Wk, ws + OFF_WKT);
  k_cvtT<<<(int)((SZ_W + 255) / 256), 256, 0, stream>>>(Wv, ws + OFF_WVT);
  k_cvt <<<(int)((SZ_P + 255) / 256), 256, 0, stream>>>(Wp, ws + OFF_WPB, (int)SZ_P);

  dim3 gQKV(T / 128, BATCH * H);
  k_qkv<<<gQKV, 256, 0, stream>>>(ws + OFF_XB, ws + OFF_WQT, ws + OFF_QB,  0);
  k_qkv<<<gQKV, 256, 0, stream>>>(ws + OFF_XB, ws + OFF_WKT, ws + OFF_KB,  0);
  k_qkv<<<gQKV, 256, 0, stream>>>(ws + OFF_XB, ws + OFF_WVT, ws + OFF_VTB, 1);

  k_attn<<<dim3(T / 128, BATCH * H), 256, 0, stream>>>(ws + OFF_QB, ws + OFF_KB,
                                                       ws + OFF_VTB, ws + OFF_YB);

  k_proj<<<dim3((BATCH * T) / 128, E / 64), 256, 0, stream>>>(ws + OFF_YB, ws + OFF_WPB,
                                                              bp, out);
}
